// RGCNN_model_11553462027010
// MI455X (gfx1250) — matrix operand order repaired and checked by reading, not compiled, not run on hardware
//
#include <hip/hip_runtime.h>
#include <math.h>

// ---------------------------------------------------------------------------
// RGCNN forward for MI455X (gfx1250): bf16 WMMA batched GEMMs, f32 accumulate.
// Round 2: register double-buffered GEMM inner loop + 4 waves per workgroup.
// ---------------------------------------------------------------------------

#define BATCH 16
#define NPTS  1024

typedef __attribute__((ext_vector_type(16))) __bf16          v16bf;
typedef __attribute__((ext_vector_type(8)))  float           v8f;
typedef __attribute__((ext_vector_type(8)))  unsigned short  us8;

union Frag { v16bf v; us8 u[2]; };

__device__ __forceinline__ unsigned short f2bf(float f) {
  union { float f; unsigned u; } x; x.f = f;
  unsigned r = x.u + 0x7FFFu + ((x.u >> 16) & 1u);   // round-to-nearest-even
  return (unsigned short)(r >> 16);
}

__device__ __forceinline__ void load_a(Frag& a, const unsigned short* arow, int k0) {
  a.u[0] = *(const us8*)(arow + k0);        // lane<16: K 0..15  | lane>=16: K 16..31
  a.u[1] = *(const us8*)(arow + k0 + 8);
}
__device__ __forceinline__ void load_b(Frag& f, const unsigned short* brow, int k0) {
  f.u[0] = *(const us8*)(brow + k0);        // lane<16: K 0..15  | lane>=16: K 16..31
  f.u[1] = *(const us8*)(brow + k0 + 8);
}

// ---------------------------------------------------------------------------
// Batched bf16 GEMM:  C[M x Nn] = A[M x K] * B[K x Nn],  B passed transposed
// (Bt is [Nn x K] row-major so each lane loads 16 contiguous K values).
// Block = 4 waves; each wave computes a 16x64 tile (4 WMMA accumulators),
// software-pipelined: next K-step fragments are in flight during the WMMAs.
// MODE 0: C = acc       MODE 1: C += acc
// MODE 2: C = 2*acc - aux[idx]          (Chebyshev recursion)
// MODE 3: C = exp(-(aux[row]+aux[col]-2*acc))   (adjacency)
// ---------------------------------------------------------------------------
template<int MODE>
__global__ __launch_bounds__(128)
void gemm_bf16(const unsigned short* __restrict__ A,
               const unsigned short* __restrict__ Bt,
               float* __restrict__ C,
               const float* __restrict__ aux,
               int M, int Nn, int K, int lda, int ldb, int ldc,
               long long sA, long long sB, long long sC, long long sAux)
{
  long long b = blockIdx.z;
  A  += b * sA;
  Bt += b * sB;
  C  += b * sC;
  const float* auxb = (MODE >= 2) ? (aux + b * sAux) : nullptr;

  const int wave = threadIdx.x >> 5;
  const int lane = threadIdx.x & 31;
  const int m0   = (blockIdx.y * 4 + wave) << 4;
  const int n0   = blockIdx.x << 6;
  const int hi   = lane >> 4;
  const int l    = lane & 15;

  v8f zero = {0.f,0.f,0.f,0.f,0.f,0.f,0.f,0.f};
  v8f acc[4] = {zero, zero, zero, zero};

  const unsigned short* arow = A  + (size_t)(m0 + l) * lda + (hi ? 16 : 0);
  const unsigned short* brow = Bt + (size_t)(n0 + l) * ldb + (hi ? 16 : 0);
  const size_t bstep = (size_t)16 * ldb;

  // pipeline prologue: fragments for k0 = 0
  Frag a0, f0[4];
  load_a(a0, arow, 0);
#pragma unroll
  for (int j = 0; j < 4; ++j) load_b(f0[j], brow + (size_t)j * bstep, 0);

  int k0 = 0;
  for (; k0 + 32 < K; k0 += 32) {
    Frag a1, f1[4];
    load_a(a1, arow, k0 + 32);                               // issue next-step loads
#pragma unroll
    for (int j = 0; j < 4; ++j) load_b(f1[j], brow + (size_t)j * bstep, k0 + 32);
#pragma unroll
    for (int j = 0; j < 4; ++j)                               // consume current step
      acc[j] = __builtin_amdgcn_wmma_f32_16x16x32_bf16(
                   false, a0.v, false, f0[j].v, (short)0, acc[j], false, false);
    a0 = a1;
#pragma unroll
    for (int j = 0; j < 4; ++j) f0[j] = f1[j];
  }
#pragma unroll
  for (int j = 0; j < 4; ++j)                                 // epilogue step
    acc[j] = __builtin_amdgcn_wmma_f32_16x16x32_bf16(
                 false, a0.v, false, f0[j].v, (short)0, acc[j], false, false);

#pragma unroll
  for (int j = 0; j < 4; ++j) {
    int col = n0 + j * 16 + l;
#pragma unroll
    for (int r = 0; r < 8; ++r) {
      int row = m0 + (hi << 3) + r;
      size_t idx = (size_t)row * ldc + col;
      float v = acc[j][r];
      if (MODE == 0)      C[idx] = v;
      else if (MODE == 1) C[idx] += v;
      else if (MODE == 2) C[idx] = 2.0f * v - auxb[idx];
      else                C[idx] = __expf(-(auxb[row] + auxb[col] - 2.0f * v));
    }
  }
}

// C[M x Nn] = sum_b A_b[M x K] * B_b   (Bt layout), batch-reduced on chip.
__global__ __launch_bounds__(128)
void gemm_bf16_reduceB(const unsigned short* __restrict__ A,
                       const unsigned short* __restrict__ Bt,
                       float* __restrict__ C,
                       int M, int Nn, int K, int lda, int ldb, int ldc,
                       long long sA, long long sB, int nB)
{
  const int wave = threadIdx.x >> 5;
  const int lane = threadIdx.x & 31;
  const int m0   = (blockIdx.y * 4 + wave) << 4;
  const int n0   = blockIdx.x << 6;
  const int hi   = lane >> 4;
  const int l    = lane & 15;

  v8f zero = {0.f,0.f,0.f,0.f,0.f,0.f,0.f,0.f};
  v8f acc[4] = {zero, zero, zero, zero};

  for (int b = 0; b < nB; ++b) {
    const unsigned short* arow = A  + (long long)b * sA + (size_t)(m0 + l) * lda + (hi ? 16 : 0);
    const unsigned short* brow = Bt + (long long)b * sB + (size_t)(n0 + l) * ldb + (hi ? 16 : 0);
    const size_t bstep = (size_t)16 * ldb;

    Frag a0, f0[4];
    load_a(a0, arow, 0);
#pragma unroll
    for (int j = 0; j < 4; ++j) load_b(f0[j], brow + (size_t)j * bstep, 0);

    int k0 = 0;
    for (; k0 + 32 < K; k0 += 32) {
      Frag a1, f1[4];
      load_a(a1, arow, k0 + 32);
#pragma unroll
      for (int j = 0; j < 4; ++j) load_b(f1[j], brow + (size_t)j * bstep, k0 + 32);
#pragma unroll
      for (int j = 0; j < 4; ++j)
        acc[j] = __builtin_amdgcn_wmma_f32_16x16x32_bf16(
                     false, a0.v, false, f0[j].v, (short)0, acc[j], false, false);
      a0 = a1;
#pragma unroll
      for (int j = 0; j < 4; ++j) f0[j] = f1[j];
    }
#pragma unroll
    for (int j = 0; j < 4; ++j)
      acc[j] = __builtin_amdgcn_wmma_f32_16x16x32_bf16(
                   false, a0.v, false, f0[j].v, (short)0, acc[j], false, false);
  }

#pragma unroll
  for (int j = 0; j < 4; ++j) {
    int col = n0 + j * 16 + l;
#pragma unroll
    for (int r = 0; r < 8; ++r) {
      int row = m0 + (hi << 3) + r;
      C[(size_t)row * ldc + col] = acc[j][r];
    }
  }
}

// ---------------------------------------------------------------------------
// Helper kernels
// ---------------------------------------------------------------------------
template<bool SQ>
__global__ void row_reduce(const float* __restrict__ X, float* __restrict__ out,
                           int rows, int C, int ld)
{
  int wid  = blockIdx.x * (blockDim.x >> 5) + (threadIdx.x >> 5);
  int lane = threadIdx.x & 31;
  if (wid >= rows) return;
  const float* r = X + (size_t)wid * ld;
  float s = 0.f;
  for (int c = lane; c < C; c += 32) { float v = r[c]; s += SQ ? v * v : v; }
#pragma unroll
  for (int o = 16; o > 0; o >>= 1) s += __shfl_xor(s, o);
  if (lane == 0) out[wid] = s;
}

__global__ void lap_prep(const float* __restrict__ rs, const float* __restrict__ W,
                         float* __restrict__ dinv, float* __restrict__ dinv0,
                         int N, int total)
{
  int i = blockIdx.x * blockDim.x + threadIdx.x;
  if (i >= total) return;
  int n = i % N;
  float s    = rs[i];
  float diag = W[(size_t)i * N + n];
  dinv[i]  = rsqrtf(fmaxf(s, 1e-20f));
  dinv0[i] = rsqrtf(fmaxf(s - diag, 1e-20f));
}

__global__ void make_cheb_lap(const float* __restrict__ W, const float* __restrict__ dinv,
                              unsigned short* __restrict__ L, int N, size_t total)
{
  size_t i = (size_t)blockIdx.x * blockDim.x + threadIdx.x;
  if (i >= total) return;
  size_t row = i / N;                 // = b*N + n
  int    m   = (int)(i - row * N);
  size_t b   = row / N;
  L[i] = f2bf(-dinv[row] * W[i] * dinv[b * N + m]);
}

__global__ void make_reg_lap(const float* __restrict__ W, const float* __restrict__ dinv0,
                             unsigned short* __restrict__ L, int N, size_t total)
{
  size_t i = (size_t)blockIdx.x * blockDim.x + threadIdx.x;
  if (i >= total) return;
  size_t row = i / N;
  int    m   = (int)(i - row * N);
  size_t b   = row / N;
  int    n   = (int)(row - b * N);
  float v = (n == m) ? 1.0f : (-dinv0[row] * W[i] * dinv0[b * N + m]);
  L[i] = f2bf(v);
}

__global__ void cvt_bf16(const float* __restrict__ in, unsigned short* __restrict__ out,
                         size_t total)
{
  size_t i = (size_t)blockIdx.x * blockDim.x + threadIdx.x;
  if (i < total) out[i] = f2bf(in[i]);
}

__global__ void transpose_cvt(const float* __restrict__ in, unsigned short* __restrict__ out,
                              int R, int C)
{
  __shared__ float tile[32][33];
  size_t base = (size_t)blockIdx.z * R * C;
  int c0 = blockIdx.x * 32, r0 = blockIdx.y * 32;
  int tx = threadIdx.x, ty = threadIdx.y;
#pragma unroll
  for (int i = 0; i < 32; i += 8)
    tile[ty + i][tx] = in[base + (size_t)(r0 + ty + i) * C + (c0 + tx)];
  __syncthreads();
#pragma unroll
  for (int i = 0; i < 32; i += 8)
    out[base + (size_t)(c0 + ty + i) * R + (r0 + tx)] = f2bf(tile[tx][ty + i]);
}

__global__ void pad_x(const float* __restrict__ x, float* __restrict__ XP,
                      int Cin, int Cpad, size_t total)
{
  size_t i = (size_t)blockIdx.x * blockDim.x + threadIdx.x;
  if (i >= total) return;
  size_t n = i / Cpad;
  int    c = (int)(i - n * Cpad);
  XP[i] = (c < Cin) ? x[n * Cin + c] : 0.0f;
}

// w layout [K,Cin,Cout] -> Wt layout [K,Cout,Kpad] bf16, zero padded over Kpad.
__global__ void transpose_w(const float* __restrict__ w, unsigned short* __restrict__ Wt,
                            int Kc, int Cin, int Kpad, int Cout)
{
  size_t total = (size_t)Kc * Cout * Kpad;
  size_t i = (size_t)blockIdx.x * blockDim.x + threadIdx.x;
  if (i >= total) return;
  size_t k   = i / ((size_t)Cout * Kpad);
  size_t rem = i - k * (size_t)Cout * Kpad;
  int o = (int)(rem / Kpad);
  int c = (int)(rem - (size_t)o * Kpad);
  Wt[i] = (c < Cin) ? f2bf(w[(k * Cin + c) * Cout + o]) : (unsigned short)0;
}

__global__ void bias_relu(float* __restrict__ X, const float* __restrict__ bias,
                          size_t total, int C)
{
  size_t i = (size_t)blockIdx.x * blockDim.x + threadIdx.x;
  if (i >= total) return;
  int c = (int)(i % C);
  X[i] = fmaxf(X[i] + bias[c], 0.0f);
}

__global__ void maxpool(const float* __restrict__ X, float* __restrict__ out, int N, int C)
{
  int b = blockIdx.y;
  int c = blockIdx.x * blockDim.x + threadIdx.x;
  if (c >= C) return;
  const float* p = X + (size_t)b * N * C + c;
  float m = -3.4e38f;
  for (int n = 0; n < N; ++n) m = fmaxf(m, p[(size_t)n * C]);
  out[(size_t)b * C + c] = m;
}

__global__ void fc(const float* __restrict__ in, const float* __restrict__ W,
                   const float* __restrict__ bias, float* __restrict__ out,
                   int I, int O, int doRelu)
{
  int b = blockIdx.y;
  int o = blockIdx.x * blockDim.x + threadIdx.x;
  if (o >= O) return;
  float s = bias[o];
  const float* xr = in + (size_t)b * I;
  for (int i = 0; i < I; ++i) s += xr[i] * W[(size_t)i * O + o];
  out[(size_t)b * O + o] = doRelu ? fmaxf(s, 0.0f) : s;
}

__global__ void sumsq(const float* __restrict__ X, float* __restrict__ accum, size_t n)
{
  size_t i = (size_t)blockIdx.x * blockDim.x + threadIdx.x;
  size_t stride = (size_t)gridDim.x * blockDim.x;
  float s = 0.f;
  for (; i < n; i += stride) { float v = X[i]; s += v * v; }
#pragma unroll
  for (int o = 16; o > 0; o >>= 1) s += __shfl_xor(s, o);
  if ((threadIdx.x & 31) == 0) atomicAdd(accum, s);
}

__global__ void zero4(float* a) { if (threadIdx.x < 4) a[threadIdx.x] = 0.0f; }

__global__ void finalize_norms(const float* __restrict__ acc, float* __restrict__ out)
{
  int i = threadIdx.x;
  if (i < 3) out[160 + i] = sqrtf(acc[i]);
}

// ---------------------------------------------------------------------------
// Host orchestration
// ---------------------------------------------------------------------------
extern "C" void kernel_launch(void* const* d_in, const int* in_sizes, int n_in,
                              void* d_out, int out_size, void* d_ws, size_t ws_size,
                              hipStream_t stream)
{
  (void)in_sizes; (void)n_in; (void)out_size; (void)ws_size;
  const float* x   = (const float*)d_in[0];
  const float* w1  = (const float*)d_in[4];
  const float* b1  = (const float*)d_in[5];
  const float* w2  = (const float*)d_in[6];
  const float* b2  = (const float*)d_in[7];
  const float* w3  = (const float*)d_in[8];
  const float* b3  = (const float*)d_in[9];
  const float* fw1 = (const float*)d_in[10];
  const float* fb1 = (const float*)d_in[11];
  const float* fw2 = (const float*)d_in[12];
  const float* fb2 = (const float*)d_in[13];
  const float* fw3 = (const float*)d_in[14];
  const float* fb3 = (const float*)d_in[15];
  float* out = (float*)d_out;

  const int B = BATCH, N = NPTS;
  char* ws = (char*)d_ws; size_t off = 0;
  auto alloc = [&](size_t bytes) -> void* {
    off = (off + 255) & ~(size_t)255;
    void* p = ws + off; off += bytes; return p;
  };

  float* Wbuf = (float*)alloc((size_t)B * N * N * 4);            // W, later M1
  unsigned short* Lb = (unsigned short*)alloc((size_t)B * N * N * 2); // Lhat, later Lreg
  float* XP   = (float*)alloc((size_t)B * N * 64 * 4);
  unsigned short* Pb  = (unsigned short*)alloc((size_t)B * N * 64 * 2);
  unsigned short* Pt  = (unsigned short*)alloc((size_t)B * N * 64 * 2);
  float* OUT1 = (float*)alloc((size_t)B * N * 128 * 4);
  float* OUT2 = (float*)alloc((size_t)B * N * 512 * 4);
  float* OUT3 = (float*)alloc((size_t)B * N * 1024 * 4);
  float* ZA   = (float*)alloc((size_t)B * N * 512 * 4);
  float* ZB   = (float*)alloc((size_t)B * N * 512 * 4);
  unsigned short* Zb  = (unsigned short*)alloc((size_t)B * N * 512 * 2);
  unsigned short* Zt  = (unsigned short*)alloc((size_t)B * N * 512 * 2);
  unsigned short* Ob1 = (unsigned short*)alloc((size_t)B * N * 128 * 2);
  unsigned short* Ot1 = (unsigned short*)alloc((size_t)B * N * 128 * 2);
  unsigned short* Ob2 = (unsigned short*)alloc((size_t)B * N * 512 * 2);
  unsigned short* Ot2 = (unsigned short*)alloc((size_t)B * N * 512 * 2);
  unsigned short* Ot3 = (unsigned short*)alloc((size_t)B * N * 1024 * 2);
  unsigned short* M1t = (unsigned short*)ZA;   // reuse: ZA (32MB) is free at reg time
  float* Mmat  = (float*)alloc((size_t)1024 * 1024 * 4);
  float* sq    = (float*)alloc((size_t)B * N * 4);
  float* rs    = (float*)alloc((size_t)B * N * 4);
  float* dinv  = (float*)alloc((size_t)B * N * 4);
  float* dinv0 = (float*)alloc((size_t)B * N * 4);
  float* pooled = (float*)alloc((size_t)B * 1024 * 4);
  float* h1 = (float*)alloc((size_t)B * 512 * 4);
  float* h2 = (float*)alloc((size_t)B * 128 * 4);
  unsigned short* Wt1 = (unsigned short*)alloc((size_t)6 * 128 * 64 * 2);
  unsigned short* Wt2 = (unsigned short*)alloc((size_t)5 * 512 * 128 * 2);
  unsigned short* Wt3 = (unsigned short*)alloc((size_t)3 * 1024 * 512 * 2);
  float* acc = (float*)alloc(4 * 4);

  auto gemm = [&](int mode, const unsigned short* A, const unsigned short* Bt,
                  float* C, const float* aux, int M, int Nn, int K,
                  int lda, int ldb, int ldc,
                  long long sA, long long sB, long long sC, long long sAux, int nB) {
    dim3 g(Nn / 64, M / 64, nB), blk(128, 1, 1);
    switch (mode) {
      case 0: gemm_bf16<0><<<g, blk, 0, stream>>>(A, Bt, C, aux, M, Nn, K, lda, ldb, ldc, sA, sB, sC, sAux); break;
      case 1: gemm_bf16<1><<<g, blk, 0, stream>>>(A, Bt, C, aux, M, Nn, K, lda, ldb, ldc, sA, sB, sC, sAux); break;
      case 2: gemm_bf16<2><<<g, blk, 0, stream>>>(A, Bt, C, aux, M, Nn, K, lda, ldb, ldc, sA, sB, sC, sAux); break;
      default: gemm_bf16<3><<<g, blk, 0, stream>>>(A, Bt, C, aux, M, Nn, K, lda, ldb, ldc, sA, sB, sC, sAux); break;
    }
  };

  auto runLayer = [&](const float* Pf, const unsigned short* Pbx, const unsigned short* Ptx,
                      float* Z0, int Cpad, int Kch, const unsigned short* Wt,
                      const float* bias, int Cout, float* OUT,
                      unsigned short* Ob, unsigned short* Ot, int regIdx) {
    const long long NN = (long long)N * N;
    const long long PC = (long long)N * Cpad;
    const long long OC = (long long)N * Cout;
    const int rows = B * N;

    // sq[b,n] = sum_c P^2
    row_reduce<true><<<dim3((rows + 7) / 8), dim3(256), 0, stream>>>(Pf, sq, rows, Cpad, Cpad);
    // W = exp(-(sq_r + sq_c - 2 P P^T))   (fused adjacency epilogue)
    gemm(3, Pbx, Pbx, Wbuf, sq, N, N, Cpad, Cpad, Cpad, N, PC, PC, NN, N, B);
    // degrees, dinv / dinv0
    row_reduce<false><<<dim3((rows + 7) / 8), dim3(256), 0, stream>>>(Wbuf, rs, rows, N, N);
    lap_prep<<<dim3((rows + 255) / 256), dim3(256), 0, stream>>>(rs, Wbuf, dinv, dinv0, N, rows);
    // Lhat (bf16, symmetric)
    {
      size_t tot = (size_t)B * NN;
      make_cheb_lap<<<dim3((unsigned)((tot + 255) / 256)), dim3(256), 0, stream>>>(Wbuf, dinv, Lb, N, tot);
    }
    // Chebyshev: out = sum_k Z_k Wk
    gemm(0, Pbx, Wt, OUT, nullptr, N, Cout, Cpad, Cpad, Cpad, Cout, PC, 0, OC, 0, B);
    if (Kch > 1) {
      const size_t totz = (size_t)B * N * Cpad;
      gemm(0, Lb, Ptx, ZA, nullptr, N, Cpad, N, N, N, Cpad, NN, PC, PC, 0, B);   // Z1 = Lhat P
      cvt_bf16<<<dim3((unsigned)((totz + 255) / 256)), dim3(256), 0, stream>>>(ZA, Zb, totz);
      transpose_cvt<<<dim3(Cpad / 32, N / 32, B), dim3(32, 8), 0, stream>>>(ZA, Zt, N, Cpad);
      gemm(1, Zb, Wt + (size_t)Cout * Cpad, OUT, nullptr, N, Cout, Cpad, Cpad, Cpad, Cout, PC, 0, OC, 0, B);
      float* zp = Z0; float* zc = ZA; float* zn = ZB;
      for (int k = 2; k < Kch; ++k) {
        gemm(2, Lb, Zt, zn, zp, N, Cpad, N, N, N, Cpad, NN, PC, PC, PC, B);       // Z2 = 2 Lhat Z1 - Z0
        cvt_bf16<<<dim3((unsigned)((totz + 255) / 256)), dim3(256), 0, stream>>>(zn, Zb, totz);
        transpose_cvt<<<dim3(Cpad / 32, N / 32, B), dim3(32, 8), 0, stream>>>(zn, Zt, N, Cpad);
        gemm(1, Zb, Wt + (size_t)k * Cout * Cpad, OUT, nullptr, N, Cout, Cpad, Cpad, Cpad, Cout, PC, 0, OC, 0, B);
        float* t = zp; zp = zc; zc = zn; zn = t;
      }
    }
    {
      size_t tot = (size_t)B * N * Cout;
      bias_relu<<<dim3((unsigned)((tot + 255) / 256)), dim3(256), 0, stream>>>(OUT, bias, tot, Cout);
      if (Ob) cvt_bf16<<<dim3((unsigned)((tot + 255) / 256)), dim3(256), 0, stream>>>(OUT, Ob, tot);
      transpose_cvt<<<dim3(Cout / 32, N / 32, B), dim3(32, 8), 0, stream>>>(OUT, Ot, N, Cout);
    }
    // reg: r = || sum_b out^T Lreg out ||_F
    {
      size_t tot = (size_t)B * NN;
      make_reg_lap<<<dim3((unsigned)((tot + 255) / 256)), dim3(256), 0, stream>>>(Wbuf, dinv0, Lb, N, tot);
      gemm(0, Lb, Ot, Wbuf, nullptr, N, Cout, N, N, N, Cout, NN, OC, OC, 0, B);   // M1 = Lreg out (into Wbuf)
      transpose_cvt<<<dim3(Cout / 32, N / 32, B), dim3(32, 8), 0, stream>>>(Wbuf, M1t, N, Cout);
      gemm_bf16_reduceB<<<dim3(Cout / 64, Cout / 64, 1), dim3(128), 0, stream>>>(
          Ot, M1t, Mmat, Cout, Cout, N, N, N, Cout, OC, OC, B);
      sumsq<<<dim3(256), dim3(256), 0, stream>>>(Mmat, acc + regIdx, (size_t)Cout * Cout);
    }
  };

  // ---- prep ----
  zero4<<<dim3(1), dim3(32), 0, stream>>>(acc);
  {
    size_t tot = (size_t)B * N * 64;
    pad_x<<<dim3((unsigned)((tot + 255) / 256)), dim3(256), 0, stream>>>(x, XP, 6, 64, tot);
    cvt_bf16<<<dim3((unsigned)((tot + 255) / 256)), dim3(256), 0, stream>>>(XP, Pb, tot);
    transpose_cvt<<<dim3(64 / 32, N / 32, B), dim3(32, 8), 0, stream>>>(XP, Pt, N, 64);
  }
  transpose_w<<<dim3((unsigned)((6u * 128 * 64 + 255) / 256)), dim3(256), 0, stream>>>(w1, Wt1, 6, 6, 64, 128);
  transpose_w<<<dim3((unsigned)((5u * 512 * 128 + 255) / 256)), dim3(256), 0, stream>>>(w2, Wt2, 5, 128, 128, 512);
  transpose_w<<<dim3((unsigned)((3u * 1024 * 512 + 255) / 256)), dim3(256), 0, stream>>>(w3, Wt3, 3, 512, 512, 1024);

  // ---- three Chebyshev graph-conv layers ----
  runLayer(XP,   Pb,  Pt,  XP,   64,  6, Wt1, b1, 128,  OUT1, Ob1,    Ot1, 0);
  runLayer(OUT1, Ob1, Ot1, OUT1, 128, 5, Wt2, b2, 512,  OUT2, Ob2,    Ot2, 1);
  runLayer(OUT2, Ob2, Ot2, OUT2, 512, 3, Wt3, b3, 1024, OUT3, nullptr, Ot3, 2);

  // ---- max-pool over points + MLP head ----
  maxpool<<<dim3(1024 / 256, B), dim3(256), 0, stream>>>(OUT3, pooled, N, 1024);
  fc<<<dim3(2, B), dim3(256), 0, stream>>>(pooled, fw1, fb1, h1, 1024, 512, 1);
  fc<<<dim3(1, B), dim3(256), 0, stream>>>(h1, fw2, fb2, h2, 512, 128, 1);
  fc<<<dim3(1, B), dim3(64), 0, stream>>>(h2, fw3, fb3, out, 128, 10, 0);

  finalize_norms<<<dim3(1), dim3(32), 0, stream>>>(acc, out);
}
